// MoE_9010841387551
// MI455X (gfx1250) — matrix operand order repaired and checked by reading, not compiled, not run on hardware
//
#include <hip/hip_runtime.h>
#include <hip/hip_bf16.h>

#define DEVI __device__ __forceinline__

namespace {

constexpr int T  = 8 * 2048;   // tokens
constexpr int E  = 1024;       // model dim
constexpr int H  = 1536;       // hidden dim
constexpr int NE = 8;          // experts
constexpr int TM = 32;         // tokens per tile

typedef __bf16 bf16x8  __attribute__((ext_vector_type(8)));
typedef __bf16 bf16x16 __attribute__((ext_vector_type(16)));
typedef float  f32x8   __attribute__((ext_vector_type(8)));

struct BfHalves { bf16x8 lo; bf16x8 hi; };
union  BfFrag   { bf16x16 v; BfHalves h; };

DEVI unsigned short f2bf(float f) {
  union { float f; unsigned int u; } c; c.f = f;
  unsigned int u = c.u;
  return (unsigned short)((u + 0x7FFFu + ((u >> 16) & 1u)) >> 16);  // RNE
}

DEVI float gelu_exact(float v) {
  return 0.5f * v * (1.0f + erff(v * 0.70710678118654752440f));
}

DEVI f32x8 wmma_bf16(bf16x16 a, bf16x16 b, f32x8 c) {
  // (neg_a, A, neg_b, B, c_mod, C, reuse_a, reuse_b)
  return __builtin_amdgcn_wmma_f32_16x16x32_bf16(false, a, false, b, (short)0, c,
                                                 false, false);
}

// A fragment (16x32 bf16, MxK) from row-major [m][ld] u16 storage.
// ISA layout: lanes 0-15 row M=lane, K={kk..kk+7, kk+16..kk+23};
//             lanes 16-31 same rows, K offset +8. Two b128 loads.
DEVI bf16x16 load_a(const unsigned short* base, int ld, int mbase, int lane, int kk) {
  const int hi = (lane >= 16) ? 8 : 0;
  const unsigned short* p = base + (size_t)(mbase + (lane & 15)) * ld + kk + hi;
  BfFrag f;
  f.h.lo = *(const bf16x8*)(p);
  f.h.hi = *(const bf16x8*)(p + 16);
  return f.v;
}

// B fragment (32x16 bf16, KxN) from [n][ldk] u16 storage (K contiguous per row).
// ISA layout: lanes 0-15 col N=lane, K={kk..kk+7, kk+16..kk+23}; lanes 16-31 K offset +8.
DEVI bf16x16 load_b(const unsigned short* ncols, int ldk, int lane, int kk) {
  const int k0 = kk + ((lane >= 16) ? 8 : 0);
  const unsigned short* p = ncols + (size_t)(lane & 15) * ldk + k0;
  BfFrag f;
  f.h.lo = *(const bf16x8*)(p);
  f.h.hi = *(const bf16x8*)(p + 16);
  return f.v;
}

DEVI void atomAddF32(float* p, float v) {
  __hip_atomic_fetch_add(p, v, __ATOMIC_RELAXED, __HIP_MEMORY_SCOPE_AGENT);
}

} // namespace

// ---------------------------------------------------------------- zero output
__global__ void moe_zero(float* __restrict__ out, int* __restrict__ counts) {
  size_t i = (size_t)blockIdx.x * blockDim.x + threadIdx.x;
  const size_t n4 = (size_t)T * E / 4;
  float4 z; z.x = z.y = z.z = z.w = 0.f;
  for (size_t k = i; k < n4; k += (size_t)gridDim.x * blockDim.x)
    ((float4*)out)[k] = z;
  if (i < NE) counts[i] = 0;
}

// --------------------------------------------- transpose + fp32 -> bf16 weights
// in: [z][R][C] f32, out: [z][C][R] bf16  (so K is contiguous per output row)
__global__ void moe_transpose_cvt(const float* __restrict__ in,
                                  unsigned short* __restrict__ outw,
                                  int R, int C) {
  __shared__ unsigned short tile[32][33];
  const float* src = in + (size_t)blockIdx.z * R * C;
  unsigned short* dst = outw + (size_t)blockIdx.z * R * C;
  const int c0 = blockIdx.x * 32, r0 = blockIdx.y * 32;
  const int tx = threadIdx.x, ty = threadIdx.y;  // 32 x 8
  for (int i = ty; i < 32; i += 8)
    tile[i][tx] = f2bf(src[(size_t)(r0 + i) * C + (c0 + tx)]);
  __syncthreads();
  for (int i = ty; i < 32; i += 8)
    dst[(size_t)(c0 + i) * R + (r0 + tx)] = tile[tx][i];
}

// ------------------------------------------------------------------- router
// One wave per token: logits = x@Wr + br, softmax, top-2, combine weights,
// append token to its two experts' lists. Also converts x row to bf16.
__global__ void moe_router(const float* __restrict__ x,
                           const float* __restrict__ Wr,
                           const float* __restrict__ br,
                           float* __restrict__ combine,
                           int* __restrict__ counts,
                           int* __restrict__ lists,
                           unsigned short* __restrict__ xbf) {
  const int wave = threadIdx.x >> 5, lane = threadIdx.x & 31;
  const int t = blockIdx.x * 8 + wave;
  const float* xr = x + (size_t)t * E;

  float acc[NE];
#pragma unroll
  for (int c = 0; c < NE; ++c) acc[c] = 0.f;

  for (int j = 0; j < E / 32; ++j) {
    const int idx = j * 32 + lane;                 // coalesced
    const float xv = xr[idx];
    xbf[(size_t)t * E + idx] = f2bf(xv);
#pragma unroll
    for (int c = 0; c < NE; ++c) acc[c] += xv * Wr[idx * NE + c];
  }
#pragma unroll
  for (int c = 0; c < NE; ++c)
    for (int off = 16; off > 0; off >>= 1)
      acc[c] += __shfl_xor(acc[c], off, 32);

  if (lane == 0) {
    float lg[NE], p[NE];
    float m = -3.4e38f;
#pragma unroll
    for (int c = 0; c < NE; ++c) { lg[c] = acc[c] + br[c]; m = fmaxf(m, lg[c]); }
    float s = 0.f;
#pragma unroll
    for (int c = 0; c < NE; ++c) { p[c] = __expf(lg[c] - m); s += p[c]; }
    const float inv = 1.0f / s;
#pragma unroll
    for (int c = 0; c < NE; ++c) p[c] *= inv;
    int i1 = 0;
#pragma unroll
    for (int c = 1; c < NE; ++c) if (p[c] > p[i1]) i1 = c;
    int i2 = (i1 == 0) ? 1 : 0;
#pragma unroll
    for (int c = 0; c < NE; ++c) if (c != i1 && p[c] > p[i2]) i2 = c;
#pragma unroll
    for (int c = 0; c < NE; ++c)
      combine[(size_t)t * NE + c] = (c == i1) ? p[i1] : ((c == i2) ? p[i2] : 0.f);
    int pos1 = atomicAdd(&counts[i1], 1);
    lists[(size_t)i1 * T + pos1] = t;
    int pos2 = atomicAdd(&counts[i2], 1);
    lists[(size_t)i2 * T + pos2] = t;
  }
}

// -------------------------------------------------------- fused expert FFN
// Per block: 32 routed tokens of one expert.
//   stage A: h = gelu(x @ W1 + b1)   (x tile in LDS, h tile to LDS)
//   stage B: out += combine * (h @ W2 + b2)  via f32 global atomics
__global__ void __launch_bounds__(256, 1)
moe_expert(const unsigned short* __restrict__ xbf,
           const unsigned short* __restrict__ w1t,   // [e][H][E] bf16
           const unsigned short* __restrict__ w2t,   // [e][E][H] bf16
           const float* __restrict__ b1,             // [e][H]
           const float* __restrict__ b2,             // [e][E]
           const int* __restrict__ counts,
           const int* __restrict__ lists,
           const float* __restrict__ combine,
           float* __restrict__ out) {
  const int e = blockIdx.y;
  const int cnt = counts[e];
  const int r0 = blockIdx.x * TM;
  if (r0 >= cnt) return;

  extern __shared__ unsigned char smem[];
  unsigned short* xs = (unsigned short*)smem;            // [TM][E]  64 KB
  unsigned short* hs = xs + (size_t)TM * E;              // [TM][H]  96 KB
  int*   tokLds = (int*)(hs + (size_t)TM * H);           // [TM]
  float* wLds   = (float*)(tokLds + TM);                 // [TM]

  const int tid = threadIdx.x, wave = tid >> 5, lane = tid & 31;

  if (tid < TM) {
    const int r = r0 + tid;
    const int tok = (r < cnt) ? lists[(size_t)e * T + r] : -1;
    tokLds[tid] = tok;
    wLds[tid] = (tok >= 0) ? combine[(size_t)tok * NE + e] : 0.f;
  }
  __syncthreads();

  // Stage x tile into LDS (zero-fill padding rows).
  for (int m = wave; m < TM; m += 8) {
    const int tok = tokLds[m];
    uint4* dst = (uint4*)(xs + (size_t)m * E);
    if (tok >= 0) {
      const uint4* src = (const uint4*)(xbf + (size_t)tok * E);
      for (int i = lane; i < E / 8; i += 32) dst[i] = src[i];
    } else {
      uint4 z; z.x = z.y = z.z = z.w = 0u;
      for (int i = lane; i < E / 8; i += 32) dst[i] = z;
    }
  }
  __syncthreads();

  const int ncol = lane & 15;
  const int mrow = (lane >= 16) ? 8 : 0;

  // ---------------- Stage A: h = gelu(x @ W1 + b1) ----------------
  const unsigned short* w1e = w1t + (size_t)e * H * E;
  for (int nt = wave; nt < H / 16; nt += 8) {            // 96 n-tiles / 8 waves
    f32x8 acc0 = {}; f32x8 acc1 = {};
    const unsigned short* bcol = w1e + (size_t)(nt * 16) * E;
#pragma unroll 2
    for (int kk = 0; kk < E; kk += 32) {
      bf16x16 b  = load_b(bcol, E, lane, kk);
      bf16x16 a0 = load_a(xs, E, 0,  lane, kk);
      bf16x16 a1 = load_a(xs, E, 16, lane, kk);
      acc0 = wmma_bf16(a0, b, acc0);
      acc1 = wmma_bf16(a1, b, acc1);
    }
    const int n = nt * 16 + ncol;
    const float bias = b1[(size_t)e * H + n];
#pragma unroll
    for (int v = 0; v < 8; ++v) {
      hs[(size_t)(v + mrow) * H + n]      = f2bf(gelu_exact(acc0[v] + bias));
      hs[(size_t)(16 + v + mrow) * H + n] = f2bf(gelu_exact(acc1[v] + bias));
    }
  }
  __syncthreads();

  // ---------------- Stage B: out += w * (h @ W2 + b2) ----------------
  const unsigned short* w2e = w2t + (size_t)e * E * H;
  for (int nt = wave; nt < E / 16; nt += 8) {            // 64 n-tiles / 8 waves
    f32x8 acc0 = {}; f32x8 acc1 = {};
    const unsigned short* bcol = w2e + (size_t)(nt * 16) * H;
#pragma unroll 2
    for (int kk = 0; kk < H; kk += 32) {
      bf16x16 b  = load_b(bcol, H, lane, kk);
      bf16x16 a0 = load_a(hs, H, 0,  lane, kk);
      bf16x16 a1 = load_a(hs, H, 16, lane, kk);
      acc0 = wmma_bf16(a0, b, acc0);
      acc1 = wmma_bf16(a1, b, acc1);
    }
    const int n = nt * 16 + ncol;
    const float bias = b2[(size_t)e * E + n];
#pragma unroll
    for (int v = 0; v < 8; ++v) {
      const int m0 = v + mrow, m1 = 16 + v + mrow;
      const int t0 = tokLds[m0], t1 = tokLds[m1];
      if (t0 >= 0) atomAddF32(out + (size_t)t0 * E + n, wLds[m0] * (acc0[v] + bias));
      if (t1 >= 0) atomAddF32(out + (size_t)t1 * E + n, wLds[m1] * (acc1[v] + bias));
    }
  }
}

// ---------------------------------------------------------------------------
extern "C" void kernel_launch(void* const* d_in, const int* in_sizes, int n_in,
                              void* d_out, int out_size, void* d_ws, size_t ws_size,
                              hipStream_t stream) {
  (void)in_sizes; (void)n_in; (void)out_size; (void)ws_size;
  const float* x  = (const float*)d_in[0];
  const float* Wr = (const float*)d_in[1];
  const float* br = (const float*)d_in[2];
  const float* W1 = (const float*)d_in[3];
  const float* b1 = (const float*)d_in[4];
  const float* W2 = (const float*)d_in[5];
  const float* b2 = (const float*)d_in[6];
  float* out = (float*)d_out;

  // Workspace carve-up (256-byte aligned regions), ~81 MB total.
  char* ws = (char*)d_ws;
  size_t off = 0;
  int* counts = (int*)(ws + off);                 off += 256;
  int* lists  = (int*)(ws + off);                 off += (size_t)NE * T * 4;
  float* comb = (float*)(ws + off);               off += (size_t)T * NE * 4;
  unsigned short* xbf = (unsigned short*)(ws + off); off += (size_t)T * E * 2;
  unsigned short* w1t = (unsigned short*)(ws + off); off += (size_t)NE * E * H * 2;
  unsigned short* w2t = (unsigned short*)(ws + off); off += (size_t)NE * E * H * 2;

  moe_zero<<<2048, 256, 0, stream>>>(out, counts);

  // W1: [e][E][H] -> [e][H][E] bf16 ; W2: [e][H][E] -> [e][E][H] bf16
  moe_transpose_cvt<<<dim3(H / 32, E / 32, NE), dim3(32, 8), 0, stream>>>(W1, w1t, E, H);
  moe_transpose_cvt<<<dim3(E / 32, H / 32, NE), dim3(32, 8), 0, stream>>>(W2, w2t, H, E);

  moe_router<<<T / 8, 256, 0, stream>>>(x, Wr, br, comb, counts, lists, xbf);

  const size_t smem = (size_t)TM * E * 2 + (size_t)TM * H * 2 + TM * 4 + TM * 4;
  moe_expert<<<dim3(T / TM, NE), 256, smem, stream>>>(xbf, w1t, w2t, b1, b2,
                                                      counts, lists, comb, out);
}
